// EquivariantThreeHopGINE_29291676958835
// MI455X (gfx1250) — matrix operand order repaired and checked by reading, not compiled, not run on hardware
//
#include <hip/hip_runtime.h>
#include <hip/hip_bf16.h>

// ---------------------------------------------------------------------------
// Types for WMMA (CDNA5 / gfx1250, wave32)
// ---------------------------------------------------------------------------
typedef __attribute__((ext_vector_type(16))) __bf16 v16bf;
typedef __attribute__((ext_vector_type(8)))  float  v8f;
typedef __attribute__((ext_vector_type(4)))  unsigned int v4u;
typedef __attribute__((ext_vector_type(4)))  float  v4f;

union BF16Frag {
    v16bf v;
    v4u   q[2];
};

__device__ __forceinline__ unsigned short f2bf(float f) {
    union { float f; unsigned u; } x;
    x.f = f;
    unsigned r = x.u + 0x7FFFu + ((x.u >> 16) & 1u);   // round-to-nearest-even
    return (unsigned short)(r >> 16);
}

// A-operand fragment from an LDS tile stored row-major [16 rows][ldk bf16].
// A 16x32 layout (matches packed B): lanes 0-15 -> K {0..15}, lanes 16-31 ->
// K {16..31}; i.e. two contiguous 8-element (16B) chunks per lane.
__device__ __forceinline__ v16bf load_afrag(const unsigned short* tile, int ldk,
                                            int lane, int kc) {
    int row  = lane & 15;
    int half = (lane >> 4) & 1;
    int base = row * ldk + kc * 32 + half * 16;
    BF16Frag u;
    u.q[0] = *(const v4u*)(tile + base);
    u.q[1] = *(const v4u*)(tile + base + 8);
    return u.v;
}

// B-operand fragment from pre-packed global weights: per (nc,kc) chunk,
// lane L holds 16 contiguous bf16 = B[kc*32 + (L>=16?16:0) + e][nc*16 + (L&15)].
__device__ __forceinline__ v16bf load_bfrag(const unsigned short* pk, int KC,
                                            int nc, int kc, int lane) {
    const v4u* p = (const v4u*)(pk + (size_t)(((nc * KC) + kc) * 32 + lane) * 16);
    BF16Frag u;
    u.q[0] = p[0];
    u.q[1] = p[1];
    return u.v;
}

__device__ __forceinline__ v8f wmma_bf16(v16bf a, v16bf b, v8f c) {
    return __builtin_amdgcn_wmma_f32_16x16x32_bf16(false, a, false, b,
                                                   (short)0, c, false, false);
}

// One 16x16 output tile: D = A(16 x KC*32) x B(KC*32 x 16), f32 accum.
__device__ __forceinline__ v8f gemm_tile(const unsigned short* atile, int ldk,
                                         const unsigned short* pk, int KC,
                                         int nc, int lane) {
    v8f acc = {};
    for (int kc = 0; kc < KC; ++kc) {
        v16bf a = load_afrag(atile, ldk, lane, kc);
        v16bf b = load_bfrag(pk, KC, nc, kc, lane);
        acc = wmma_bf16(a, b, acc);
    }
    return acc;
}

// ---------------------------------------------------------------------------
// Weight pre-pack: f32 row-major W[N][K]  ->  bf16 WMMA-B layout (padded)
// ---------------------------------------------------------------------------
__global__ void pack_b_kernel(const float* __restrict__ W, int K, int N,
                              int Kp, int Np, unsigned short* __restrict__ dst) {
    int total = Kp * Np;
    int t = blockIdx.x * blockDim.x + threadIdx.x;
    if (t >= total) return;
    int KC    = Kp >> 5;
    int e     = t & 15;
    int lane  = (t >> 4) & 31;
    int chunk = t >> 9;             // nc*KC + kc
    int kc = chunk % KC;
    int nc = chunk / KC;
    int n = nc * 16 + (lane & 15);
    int k = kc * 32 + ((lane >> 4) & 1) * 16 + e;
    float v = (k < K && n < N) ? W[(size_t)n * K + k] : 0.0f;
    dst[t] = f2bf(v);
}

// ---------------------------------------------------------------------------
// Edge-type bias precompute: ebias[l][t][0..127] = lin_e_l(bond[t]) (padded)
// ---------------------------------------------------------------------------
__global__ void ebias_kernel(const float* __restrict__ W1, const float* __restrict__ b1,
                             const float* __restrict__ W2, const float* __restrict__ b2,
                             const float* __restrict__ W3, const float* __restrict__ b3,
                             const float* __restrict__ bond,
                             float* __restrict__ ebias) {
    int t = blockIdx.x * blockDim.x + threadIdx.x;
    if (t >= 3 * 5 * 128) return;
    int l  = t / 640;
    int r  = t % 640;
    int et = r >> 7;
    int n  = r & 127;
    const float* W = (l == 0) ? W1 : ((l == 1) ? W2 : W3);
    const float* b = (l == 0) ? b1 : ((l == 1) ? b2 : b3);
    int dim = (l == 0) ? 120 : 128;
    float v = 0.0f;
    if (n < dim) {
        v = b[n];
        for (int k = 0; k < 32; ++k) v += W[n * 32 + k] * bond[et * 32 + k];
    }
    ebias[t] = v;
}

// ---------------------------------------------------------------------------
// Atom embedding: 27 table gathers -> h0 [N][128] (cols 120..127 zero pad)
// ---------------------------------------------------------------------------
struct AtomTabs { const float* p[27]; };

__global__ void atom_embed_kernel(const int* __restrict__ feats, AtomTabs tabs,
                                  float* __restrict__ h0, int N) {
    int n = blockIdx.x * blockDim.x + threadIdx.x;
    if (n >= N) return;
    size_t ob = (size_t)n * 128;
    {
        int idx = feats[(size_t)n * 27 + 0];
        const float* tp = tabs.p[0] + (size_t)idx * 16;
        for (int j = 0; j < 16; ++j) h0[ob + j] = tp[j];
    }
    for (int i = 1; i < 27; ++i) {
        int idx = feats[(size_t)n * 27 + i] + (i == 2 ? 1 : 0);  // valence +1
        const float* tp = tabs.p[i] + (size_t)idx * 4;
        int co = 16 + 4 * (i - 1);
        for (int j = 0; j < 4; ++j) h0[ob + co + j] = tp[j];
    }
    for (int j = 120; j < 128; ++j) h0[ob + j] = 0.0f;
}

// ---------------------------------------------------------------------------
// agg init: agg = x   (vectorized 16B copy)
// ---------------------------------------------------------------------------
__global__ void copy4_kernel(const float* __restrict__ src, float* __restrict__ dst,
                             size_t n4) {
    size_t t = (size_t)blockIdx.x * blockDim.x + threadIdx.x;
    if (t < n4) ((v4f*)dst)[t] = ((const v4f*)src)[t];
}

// ---------------------------------------------------------------------------
// Edge scatter: agg[dst] += relu(x[src] + ebias[type]); symmetrized edges.
// One thread = one edge direction x 4 channels. L2-resident f32 atomics.
// ---------------------------------------------------------------------------
__global__ void edge_scatter_kernel(const int* __restrict__ srcI,
                                    const int* __restrict__ dstI,
                                    const int* __restrict__ ebI,
                                    const float* __restrict__ x,
                                    const float* __restrict__ ebias,  // [5][128]
                                    float* __restrict__ agg, int E) {
    long long g = (long long)blockIdx.x * blockDim.x + threadIdx.x;
    long long total = 2LL * E * 32;
    if (g >= total) return;
    int cg = (int)(g & 31);
    long long eidx = g >> 5;
    int fwd = eidx < E;
    int e = fwd ? (int)eidx : (int)(eidx - E);
    int s = fwd ? srcI[e] : dstI[e];
    int d = fwd ? dstI[e] : srcI[e];
    int t = ebI[e];
    t = (t >= 1 && t <= 4) ? t : 0;
    v4f xv = *(const v4f*)(x + (size_t)s * 128 + cg * 4);
    v4f bv = *(const v4f*)(ebias + t * 128 + cg * 4);
    float m0 = xv.x + bv.x; m0 = m0 > 0.f ? m0 : 0.f;
    float m1 = xv.y + bv.y; m1 = m1 > 0.f ? m1 : 0.f;
    float m2 = xv.z + bv.z; m2 = m2 > 0.f ? m2 : 0.f;
    float m3 = xv.w + bv.w; m3 = m3 > 0.f ? m3 : 0.f;
    float* ap = agg + (size_t)d * 128 + cg * 4;
    atomicAdd(ap + 0, m0);
    atomicAdd(ap + 1, m1);
    atomicAdd(ap + 2, m2);
    atomicAdd(ap + 3, m3);
}

// ---------------------------------------------------------------------------
// Fused GINE node MLP: h = LN( relu(fc2(relu(fc1(agg)))) * res + skip )
// One wave per 16-node tile, 2 waves/block. All GEMMs via v_wmma bf16.
// skip: layer 1 -> h0 @ skip0_W^T (WMMA); layers 2/3 -> x directly.
// ---------------------------------------------------------------------------
#define NODE_WAVE_LDS 20608
__global__ __launch_bounds__(64) void gine_node_kernel(
    const float* __restrict__ xin,                 // agg [N][128]
    const float* __restrict__ skipsrc,             // h0 (layer1) or nullptr
    const unsigned short* __restrict__ skipWpk,    // packed skip0 or nullptr
    const float* __restrict__ xdirect,             // h_{l-1} (layers 2/3) or null
    const unsigned short* __restrict__ fc1pk, const float* __restrict__ fc1b,
    const unsigned short* __restrict__ fc2pk, const float* __restrict__ fc2b,
    const float* __restrict__ lng, const float* __restrict__ lnb,
    const float* __restrict__ resp,
    float* __restrict__ hout) {
    __shared__ __align__(16) unsigned char smem[2 * NODE_WAVE_LDS];
    int lane = threadIdx.x & 31;
    int wave = threadIdx.x >> 5;
    unsigned char* wb = smem + wave * NODE_WAVE_LDS;
    unsigned short* Xs   = (unsigned short*)(wb);            // 16x128 bf16 = 4KB
    unsigned short* Ss   = (unsigned short*)(wb + 4096);     // 16x128 bf16 = 4KB
    unsigned short* Ms   = (unsigned short*)(wb + 8192);     // 16x128 bf16 = 4KB
    float*          Os   = (float*)(wb + 12288);             // 16x128 f32  = 8KB
    float*          Stat = (float*)(wb + 20480);             // 16x{mu,rs}

    int tile = blockIdx.x * 2 + wave;
    int m0 = tile * 16;
    float res = resp[0];

    // Stage input tile (and skip source for layer 1) as bf16, row-major.
    for (int i = lane; i < 16 * 128; i += 32) {
        int r = i >> 7, c = i & 127;
        Xs[i] = f2bf(xin[(size_t)(m0 + r) * 128 + c]);
        if (skipWpk) Ss[i] = f2bf(skipsrc[(size_t)(m0 + r) * 128 + c]);
    }
    __syncthreads();

    // fc1: 16x128 @ 128x128, relu, -> Ms (bf16)
    for (int nc = 0; nc < 8; ++nc) {
        v8f d = gemm_tile(Xs, 128, fc1pk, 4, nc, lane);
        int col = nc * 16 + (lane & 15);
        float bb = fc1b[col];
        for (int r = 0; r < 8; ++r) {
            float v = d[r] + bb;
            v = v > 0.f ? v : 0.f;
            int M = r + 8 * (lane >> 4);
            Ms[M * 128 + col] = f2bf(v);
        }
    }
    __syncthreads();

    // skip term into Os
    if (skipWpk) {
        for (int nc = 0; nc < 8; ++nc) {
            v8f s = gemm_tile(Ss, 128, skipWpk, 4, nc, lane);
            int col = nc * 16 + (lane & 15);
            for (int r = 0; r < 8; ++r)
                Os[(r + 8 * (lane >> 4)) * 128 + col] = s[r];
        }
    } else {
        for (int i = lane; i < 2048; i += 32) {
            int r = i >> 7, c = i & 127;
            Os[i] = xdirect[(size_t)(m0 + r) * 128 + c];
        }
    }
    __syncthreads();

    // fc2: relu(.)*res + skip  -> Os
    for (int nc = 0; nc < 8; ++nc) {
        v8f d = gemm_tile(Ms, 128, fc2pk, 4, nc, lane);
        int col = nc * 16 + (lane & 15);
        float bb = fc2b[col];
        for (int r = 0; r < 8; ++r) {
            float v = d[r] + bb;
            v = v > 0.f ? v : 0.f;
            int M = r + 8 * (lane >> 4);
            Os[M * 128 + col] += v * res;
        }
    }
    __syncthreads();

    // LayerNorm over 128 features per node
    if (lane < 16) {
        float s = 0.f, s2 = 0.f;
        for (int c = 0; c < 128; ++c) {
            float v = Os[lane * 128 + c];
            s += v; s2 += v * v;
        }
        float mu = s * (1.0f / 128.0f);
        float var = s2 * (1.0f / 128.0f) - mu * mu;
        Stat[lane * 2]     = mu;
        Stat[lane * 2 + 1] = rsqrtf(var + 1e-5f);
    }
    __syncthreads();
    for (int i = lane; i < 2048; i += 32) {
        int r = i >> 7, c = i & 127;
        float y = (Os[i] - Stat[r * 2]) * Stat[r * 2 + 1] * lng[c] + lnb[c];
        hout[(size_t)(m0 + r) * 128 + c] = y;
    }
}

// ---------------------------------------------------------------------------
// Fused head: out = out_lin( relu(mix2( relu(mix1( [h0|h1|h2|h3] )))))
// hcat K = 504 -> 512; all three GEMMs via WMMA, activations kept in LDS.
// ---------------------------------------------------------------------------
#define HEAD_WAVE_LDS 24576
__global__ __launch_bounds__(64) void head_kernel(
    const float* __restrict__ h0, const float* __restrict__ h1,
    const float* __restrict__ h2, const float* __restrict__ h3,
    const unsigned short* __restrict__ mix1pk, const float* __restrict__ mix1b,
    const unsigned short* __restrict__ mix2pk, const float* __restrict__ mix2b,
    const unsigned short* __restrict__ outpk,  const float* __restrict__ outb,
    float* __restrict__ out) {
    __shared__ __align__(16) unsigned char smem[2 * HEAD_WAVE_LDS];
    int lane = threadIdx.x & 31;
    int wave = threadIdx.x >> 5;
    unsigned char* wb = smem + wave * HEAD_WAVE_LDS;
    unsigned short* Acat = (unsigned short*)(wb);            // 16x512 bf16 = 16KB
    unsigned short* Mid1 = (unsigned short*)(wb + 16384);    // 16x256 bf16 =  8KB
    unsigned short* Mid2 = Acat;                             // reuse (Acat dead)

    int tile = blockIdx.x * 2 + wave;
    int m0 = tile * 16;

    for (int i = lane; i < 16 * 512; i += 32) {
        int r = i >> 9, k = i & 511;
        size_t rb = (size_t)(m0 + r) * 128;
        float v;
        if (k < 120)      v = h0[rb + k];
        else if (k < 248) v = h1[rb + (k - 120)];
        else if (k < 376) v = h2[rb + (k - 248)];
        else if (k < 504) v = h3[rb + (k - 376)];
        else              v = 0.0f;
        Acat[i] = f2bf(v);
    }
    __syncthreads();

    // mix1: 16x512 @ 512x256, relu -> Mid1
    for (int nc = 0; nc < 16; ++nc) {
        v8f d = gemm_tile(Acat, 512, mix1pk, 16, nc, lane);
        int col = nc * 16 + (lane & 15);
        float bb = mix1b[col];
        for (int r = 0; r < 8; ++r) {
            float v = d[r] + bb;
            v = v > 0.f ? v : 0.f;
            Mid1[(r + 8 * (lane >> 4)) * 256 + col] = f2bf(v);
        }
    }
    __syncthreads();

    // mix2: 16x256 @ 256x128, relu -> Mid2 (reuses Acat region)
    for (int nc = 0; nc < 8; ++nc) {
        v8f d = gemm_tile(Mid1, 256, mix2pk, 8, nc, lane);
        int col = nc * 16 + (lane & 15);
        float bb = mix2b[col];
        for (int r = 0; r < 8; ++r) {
            float v = d[r] + bb;
            v = v > 0.f ? v : 0.f;
            Mid2[(r + 8 * (lane >> 4)) * 128 + col] = f2bf(v);
        }
    }
    __syncthreads();

    // out: 16x128 @ 128x120 (N padded to 128), store tight [N][120]
    for (int nc = 0; nc < 8; ++nc) {
        v8f d = gemm_tile(Mid2, 128, outpk, 4, nc, lane);
        int col = nc * 16 + (lane & 15);
        if (col < 120) {
            float bb = outb[col];
            for (int r = 0; r < 8; ++r) {
                int M = r + 8 * (lane >> 4);
                out[(size_t)(m0 + M) * 120 + col] = d[r] + bb;
            }
        }
    }
}

// ---------------------------------------------------------------------------
// kernel_launch
// ---------------------------------------------------------------------------
extern "C" void kernel_launch(void* const* d_in, const int* in_sizes, int n_in,
                              void* d_out, int out_size, void* d_ws, size_t ws_size,
                              hipStream_t stream) {
    const int* features = (const int*)d_in[0];
    const int* srcI     = (const int*)d_in[1];
    const int* dstI     = (const int*)d_in[2];
    const int* ebI      = (const int*)d_in[3];
    AtomTabs tabs;
    for (int i = 0; i < 27; ++i) tabs.p[i] = (const float*)d_in[4 + i];
    const float* bond = (const float*)d_in[31];
    const float* g_leW[3] = { (const float*)d_in[32], (const float*)d_in[38], (const float*)d_in[44] };
    const float* g_leB[3] = { (const float*)d_in[33], (const float*)d_in[39], (const float*)d_in[45] };
    const float* g_f1W[3] = { (const float*)d_in[34], (const float*)d_in[40], (const float*)d_in[46] };
    const float* g_f1B[3] = { (const float*)d_in[35], (const float*)d_in[41], (const float*)d_in[47] };
    const float* g_f2W[3] = { (const float*)d_in[36], (const float*)d_in[42], (const float*)d_in[48] };
    const float* g_f2B[3] = { (const float*)d_in[37], (const float*)d_in[43], (const float*)d_in[49] };
    const float* lnG[3]   = { (const float*)d_in[50], (const float*)d_in[52], (const float*)d_in[54] };
    const float* lnB[3]   = { (const float*)d_in[51], (const float*)d_in[53], (const float*)d_in[55] };
    const float* resP[3]  = { (const float*)d_in[56], (const float*)d_in[57], (const float*)d_in[58] };
    const float* skip0W   = (const float*)d_in[59];
    const float* mix1W = (const float*)d_in[60]; const float* mix1B = (const float*)d_in[61];
    const float* mix2W = (const float*)d_in[62]; const float* mix2B = (const float*)d_in[63];
    const float* outW  = (const float*)d_in[64]; const float* outB  = (const float*)d_in[65];

    const int N = in_sizes[0] / 27;   // 131072
    const int E = in_sizes[1];        // 262144

    // Workspace layout (bytes). Needs ~337 MB.
    char* w = (char*)d_ws;
    const size_t hbytes = (size_t)N * 128 * sizeof(float);   // 64 MB each
    float* h0  = (float*)(w);
    float* h1  = (float*)(w + hbytes);
    float* h2  = (float*)(w + 2 * hbytes);
    float* h3  = (float*)(w + 3 * hbytes);
    float* agg = (float*)(w + 4 * hbytes);
    float* ebias = (float*)(w + 5 * hbytes);                 // 3*5*128 f32
    unsigned short* pk = (unsigned short*)(w + 5 * hbytes + 7680);
    unsigned short* pk_skip0 = pk;                  // 128x128
    unsigned short* pk_f1[3] = { pk + 16384, pk + 2 * 16384, pk + 3 * 16384 };
    unsigned short* pk_f2[3] = { pk + 4 * 16384, pk + 5 * 16384, pk + 6 * 16384 };
    unsigned short* pk_mix1 = pk + 7 * 16384;       // 512x256
    unsigned short* pk_mix2 = pk_mix1 + 131072;     // 256x128
    unsigned short* pk_out  = pk_mix2 + 32768;      // 128x128

    // 1) Atom embedding -> h0 (padded to 128)
    atom_embed_kernel<<<(N + 255) / 256, 256, 0, stream>>>(features, tabs, h0, N);

    // 2) Pack all GEMM weights into bf16 WMMA-B layout
    pack_b_kernel<<<(128 * 128 + 255) / 256, 256, 0, stream>>>(skip0W, 120, 128, 128, 128, pk_skip0);
    pack_b_kernel<<<(128 * 128 + 255) / 256, 256, 0, stream>>>(g_f1W[0], 120, 128, 128, 128, pk_f1[0]);
    pack_b_kernel<<<(128 * 128 + 255) / 256, 256, 0, stream>>>(g_f2W[0], 128, 128, 128, 128, pk_f2[0]);
    pack_b_kernel<<<(128 * 128 + 255) / 256, 256, 0, stream>>>(g_f1W[1], 128, 128, 128, 128, pk_f1[1]);
    pack_b_kernel<<<(128 * 128 + 255) / 256, 256, 0, stream>>>(g_f2W[1], 128, 128, 128, 128, pk_f2[1]);
    pack_b_kernel<<<(128 * 128 + 255) / 256, 256, 0, stream>>>(g_f1W[2], 128, 128, 128, 128, pk_f1[2]);
    pack_b_kernel<<<(128 * 128 + 255) / 256, 256, 0, stream>>>(g_f2W[2], 128, 128, 128, 128, pk_f2[2]);
    pack_b_kernel<<<(512 * 256 + 255) / 256, 256, 0, stream>>>(mix1W, 504, 256, 512, 256, pk_mix1);
    pack_b_kernel<<<(256 * 128 + 255) / 256, 256, 0, stream>>>(mix2W, 256, 128, 256, 128, pk_mix2);
    pack_b_kernel<<<(128 * 128 + 255) / 256, 256, 0, stream>>>(outW, 128, 120, 128, 128, pk_out);

    // 3) Per-edge-type message biases: lin_e_l(bond[t]) for t in 0..4
    ebias_kernel<<<(3 * 5 * 128 + 255) / 256, 256, 0, stream>>>(
        g_leW[0], g_leB[0], g_leW[1], g_leB[1], g_leW[2], g_leB[2], bond, ebias);

    const size_t n4 = (size_t)N * 128 / 4;
    const int copyBlocks = (int)((n4 + 255) / 256);
    const long long eThreads = 2LL * E * 32;
    const int edgeBlocks = (int)((eThreads + 255) / 256);
    const int nodeBlocks = N / 32;     // 2 tiles (waves) per 64-thread block

    const float* layerIn[3]  = { h0, h1, h2 };
    float*       layerOut[3] = { h1, h2, h3 };

    for (int l = 0; l < 3; ++l) {
        // agg = x
        copy4_kernel<<<copyBlocks, 256, 0, stream>>>(layerIn[l], agg, n4);
        // agg[dst] += relu(x[src] + ebias[type])  (both directions)
        edge_scatter_kernel<<<edgeBlocks, 256, 0, stream>>>(
            srcI, dstI, ebI, layerIn[l], ebias + l * 640, agg, E);
        // h_{l+1} = LN(relu(fc2(relu(fc1(agg)))) * res + skip)
        gine_node_kernel<<<nodeBlocks, 64, 0, stream>>>(
            agg,
            (l == 0) ? h0 : nullptr,
            (l == 0) ? pk_skip0 : nullptr,
            (l == 0) ? nullptr : layerIn[l],
            pk_f1[l], g_f1B[l], pk_f2[l], g_f2B[l],
            lnG[l], lnB[l], resP[l], layerOut[l]);
    }

    // 4) Head: mix1 -> mix2 -> out (all WMMA, fused)
    head_kernel<<<nodeBlocks, 64, 0, stream>>>(
        h0, h1, h2, h3, pk_mix1, mix1B, pk_mix2, mix2B, pk_out, outB,
        (float*)d_out);
}
